// RegMemFewShotModel_6906307412403
// MI455X (gfx1250) — matrix operand order repaired and checked by reading, not compiled, not run on hardware
//
#include <hip/hip_runtime.h>
#include <hip/hip_bf16.h>
#include <math.h>

// Flip to 0 if the async-to-LDS inline asm is rejected by the assembler.
#ifndef CDNA5_ASYNC
#define CDNA5_ASYNC 1
#endif

typedef __attribute__((ext_vector_type(16))) _Float16 v16h;
typedef __attribute__((ext_vector_type(8)))  float    v8f;
typedef __attribute__((ext_vector_type(4)))  float    v4f;
typedef __attribute__((ext_vector_type(4)))  unsigned int v4u;

#define F_BIG 3.0e38f
#define EPS_F 1e-12f

// ---- s_wait_asynccnt <= N ------------------------------------------------
template <int N>
__device__ __forceinline__ void async_wait() {
#if CDNA5_ASYNC
#if __has_builtin(__builtin_amdgcn_s_wait_asynccnt)
  __builtin_amdgcn_s_wait_asynccnt(N);
#else
  asm volatile("s_wait_asynccnt %0" ::"n"(N) : "memory");
#endif
#endif
}

// -------------------------------------------------------------------------
// prep_mem: mem [M,C] fp32 -> mh [M,C] f16, mn[M] = ||m||^2
// -------------------------------------------------------------------------
template <int C>
__global__ void __launch_bounds__(256)
prep_mem_kernel(const float* __restrict__ mem, _Float16* __restrict__ mh,
                float* __restrict__ mn, int M) {
  const int wave = threadIdx.x >> 5;
  const int lane = threadIdx.x & 31;
  const int row  = blockIdx.x * 8 + wave;
  if (row >= M) return;
  float s = 0.f;
#pragma unroll
  for (int c0 = 0; c0 < C; c0 += 32) {
    const int c = c0 + lane;
    const float v = mem[row * C + c];
    s += v * v;
    mh[row * C + c] = (_Float16)v;
  }
#pragma unroll
  for (int off = 16; off > 0; off >>= 1) s += __shfl_down(s, off, 32);
  if (lane == 0) mn[row] = s;
}

// -------------------------------------------------------------------------
// prep_q_maha: q [B,C,H,W] fp32 -> qh [N,C] f16, qn[N] = ||q||^2,
//              maha[N] = sqrt(max(diff^T icov diff, eps)), diff = q - mu
// -------------------------------------------------------------------------
template <int C>
__global__ void __launch_bounds__(256)
prep_q_maha_kernel(const float* __restrict__ q, const float* __restrict__ mu,
                   const float* __restrict__ icov, _Float16* __restrict__ qh,
                   float* __restrict__ qn, float* __restrict__ maha,
                   int Npix, int HW) {
  constexpr int P  = 32;
  constexpr int KG = 256 / P;
  __shared__ float tile[P][C + 1];
  __shared__ float psum[KG][P];

  const int nBase = blockIdx.x * P;

  for (int idx = threadIdx.x; idx < P * C; idx += 256) {
    const int c = idx / P;
    const int p = idx % P;
    const int n = nBase + p;
    float v = 0.f;
    if (n < Npix) {
      const int b  = n / HW;
      const int hw = n % HW;
      v = q[(b * C + c) * HW + hw];
    }
    tile[p][c] = v;
  }
  __syncthreads();

  for (int idx = threadIdx.x; idx < P * C; idx += 256) {
    const int p = idx / C;
    const int c = idx % C;
    const int n = nBase + p;
    if (n < Npix) qh[n * C + c] = (_Float16)tile[p][c];
  }
  if (threadIdx.x < P) {
    const int p = threadIdx.x;
    const int n = nBase + p;
    float s = 0.f;
    for (int c = 0; c < C; ++c) {
      const float v = tile[p][c];
      s += v * v;
    }
    if (n < Npix) qn[n] = s;
  }
  __syncthreads();

  for (int idx = threadIdx.x; idx < P * C; idx += 256) {
    const int c = idx / P;
    const int p = idx % P;
    tile[p][c] -= mu[c];
  }
  __syncthreads();

  {
    const int kq = threadIdx.x / P;
    const int p  = threadIdx.x % P;
    float s = 0.f;
    for (int k = kq; k < C; k += KG) {
      const float dk = tile[p][k];
      const float* __restrict__ irow = icov + k * C;
      float inner = 0.f;
      for (int j = 0; j < C; ++j) inner += irow[j] * tile[p][j];
      s += dk * inner;
    }
    psum[kq][p] = s;
  }
  __syncthreads();
  if (threadIdx.x < P) {
    const int p = threadIdx.x;
    const int n = nBase + p;
    float tot = 0.f;
#pragma unroll
    for (int g = 0; g < KG; ++g) tot += psum[g][p];
    if (n < Npix) maha[n] = sqrtf(fmaxf(tot, EPS_F));
  }
}

// -------------------------------------------------------------------------
// Stage one 32-row mem tile (rows mb..mb+31) into LDS buffer `buf`.
// LDS row stride = C+8 halfs (bank-conflict padding).
// -------------------------------------------------------------------------
template <int C>
__device__ __forceinline__ void stage_issue(const _Float16* __restrict__ mh,
                                            char* smem, int buf, int mb,
                                            int tid) {
  constexpr int RSTR = C + 8;      // halfs per LDS row
  constexpr int CPR  = C / 8;      // 16B chunks per row
  constexpr int CPT  = C / 32;     // chunks per thread (128 threads, 32 rows)
  const unsigned bufBase =
      (unsigned)(uintptr_t)smem + (unsigned)(buf * 32 * RSTR * 2);
#pragma unroll
  for (int k = 0; k < CPT; ++k) {
    const int chunk = tid + k * 128;
    const int row   = chunk / CPR;
    const int colh  = (chunk % CPR) * 8;
#if CDNA5_ASYNC
    const unsigned lds  = bufBase + (unsigned)(row * RSTR + colh) * 2u;
    const unsigned voff = (unsigned)((mb + row) * C + colh) * 2u;
    asm volatile("global_load_async_to_lds_b128 %0, %1, %2"
                 ::"v"(lds), "v"(voff),
                   "s"((unsigned long long)(uintptr_t)mh)
                 : "memory");
#else
    const v4u val = *(const v4u*)(mh + (size_t)(mb + row) * C + colh);
    *(v4u*)(smem + (size_t)buf * 32 * RSTR * 2 + (row * RSTR + colh) * 2) = val;
#endif
  }
}

// -------------------------------------------------------------------------
// knn: fused distance GEMM (WMMA f16->f32) + streaming top-5 min per row.
// block = 128 threads = 4 waves; block owns 64 rows (16 per wave); all waves
// share one double-buffered 32-column B stage in LDS (async producer).
// -------------------------------------------------------------------------
template <int C>
__global__ void __launch_bounds__(128)
knn_kernel(const _Float16* __restrict__ qh, const _Float16* __restrict__ mh,
           const float* __restrict__ qn, const float* __restrict__ mn,
           const float* __restrict__ maha, float* __restrict__ m_out, int M) {
  constexpr int KFRAG = C / 32;
  constexpr int RSTR  = C + 8;
  constexpr int STAGE_BYTES = 2 * 32 * RSTR * 2;
  constexpr int CAND_BYTES  = 4 * 16 * 16 * 5 * 4;
  constexpr int SMEM_BYTES  = STAGE_BYTES > CAND_BYTES ? STAGE_BYTES : CAND_BYTES;
  constexpr int CPT = C / 32;
  __shared__ __align__(16) char smem[SMEM_BYTES];

  const int tid  = threadIdx.x;
  const int wave = tid >> 5;
  const int lane = tid & 31;
  const int l16  = lane & 15;
  const int hi   = lane >> 4;
  const int rBase = blockIdx.x * 64 + wave * 16;  // 16 rows per wave

  // ---- A fragments, preloaded for the whole M sweep ----
  // CDNA5 16-bit A 16x32 layout: lanes 0-15 -> row l, K {0..7,16..23};
  //                              lanes 16-31 -> row l-16, K {8..15,24..31}
  v16h afrag[KFRAG];
  {
    const int arow = rBase + l16;
#pragma unroll
    for (int kb = 0; kb < C; kb += 32) {
      const _Float16* base = qh + (size_t)arow * C + kb + hi * 8;
      union { v16h h; v4f f[2]; } u;
      u.f[0] = *(const v4f*)(base);
      u.f[1] = *(const v4f*)(base + 16);
      afrag[kb / 32] = u.h;
    }
  }
  float qnr[8];
#pragma unroll
  for (int i = 0; i < 8; ++i) qnr[i] = qn[rBase + hi * 8 + i];

  float t5[8][5];
#pragma unroll
  for (int i = 0; i < 8; ++i)
#pragma unroll
    for (int j = 0; j < 5; ++j) t5[i][j] = F_BIG;

  // ---- double-buffered sweep over M in 32-column stages ----
  stage_issue<C>(mh, smem, 0, 0, tid);

  for (int mb = 0; mb < M; mb += 32) {
    const int buf = (mb >> 5) & 1;
    __syncthreads();  // previous consumers of buf^1 are done -> safe to refill
    if (mb + 32 < M) {
      stage_issue<C>(mh, smem, buf ^ 1, mb + 32, tid);
      async_wait<CPT>();  // retire current buffer's loads (next batch in flight)
    } else {
      async_wait<0>();
    }
    __syncthreads();  // all waves observed the staged data

    const _Float16* bs = (const _Float16*)smem + (size_t)buf * 32 * RSTR;
    v8f acc0 = {}, acc1 = {};
#pragma unroll
    for (int kb = 0; kb < C; kb += 32) {
      // CDNA5 16-bit B 32x16 layout: lanes 0-15 -> col l, K {0..7,16..23};
      //                              lanes 16-31 -> col l-16, K {8..15,24..31}
      const _Float16* p0 = bs + l16 * RSTR + kb + hi * 8;
      union { v16h h; v4f f[2]; } u0, u1;
      u0.f[0] = *(const v4f*)(p0);
      u0.f[1] = *(const v4f*)(p0 + 16);
      const _Float16* p1 = p0 + 16 * RSTR;
      u1.f[0] = *(const v4f*)(p1);
      u1.f[1] = *(const v4f*)(p1 + 16);
      // two independent accumulator chains -> WMMAs pipeline
      acc0 = __builtin_amdgcn_wmma_f32_16x16x32_f16(
          false, afrag[kb / 32], false, u0.h, (short)0, acc0, false, false);
      acc1 = __builtin_amdgcn_wmma_f32_16x16x32_f16(
          false, afrag[kb / 32], false, u1.h, (short)0, acc1, false, false);
    }
    const float mnc0 = mn[mb + l16];
    const float mnc1 = mn[mb + 16 + l16];
#pragma unroll
    for (int i = 0; i < 8; ++i) {
      const float d2a = qnr[i] + mnc0 - 2.0f * acc0[i];
      if (d2a < t5[i][4]) {
        t5[i][4] = d2a;
#pragma unroll
        for (int j = 4; j >= 1; --j) {
          const float lo = t5[i][j - 1], hj = t5[i][j];
          t5[i][j - 1] = fminf(lo, hj);
          t5[i][j]     = fmaxf(lo, hj);
        }
      }
      const float d2b = qnr[i] + mnc1 - 2.0f * acc1[i];
      if (d2b < t5[i][4]) {
        t5[i][4] = d2b;
#pragma unroll
        for (int j = 4; j >= 1; --j) {
          const float lo = t5[i][j - 1], hj = t5[i][j];
          t5[i][j - 1] = fminf(lo, hj);
          t5[i][j]     = fmaxf(lo, hj);
        }
      }
    }
  }

  // ---- per-wave merge: 16 lane-columns x 5 candidates per row ----
  __syncthreads();  // all waves done reading stage -> safe to alias with cand
  float* cand = (float*)smem + wave * (16 * 16 * 5);
#pragma unroll
  for (int i = 0; i < 8; ++i)
#pragma unroll
    for (int j = 0; j < 5; ++j)
      cand[((hi * 8 + i) * 16 + l16) * 5 + j] = t5[i][j];
  __syncthreads();

  if (lane < 16) {
    const int r = lane;
    float best[5];
#pragma unroll
    for (int j = 0; j < 5; ++j) best[j] = F_BIG;
    for (int l = 0; l < 16; ++l)
#pragma unroll
      for (int j = 0; j < 5; ++j) {
        float v = cand[(r * 16 + l) * 5 + j];
        if (v >= best[4]) break;  // candidate lists sorted ascending
#pragma unroll
        for (int x = 0; x < 5; ++x) {
          const float cur = best[x];
          best[x] = fminf(cur, v);
          v       = fmaxf(cur, v);
        }
      }
    float s = 0.f;
#pragma unroll
    for (int j = 0; j < 5; ++j) s += sqrtf(fmaxf(best[j], EPS_F));
    const float knn = s * 0.2f;
    const int n = rBase + r;
    m_out[n] = 0.5f * knn + 0.5f * maha[n];
  }
}

// -------------------------------------------------------------------------
// final: combined map + per-batch top-10 mean
// -------------------------------------------------------------------------
__global__ void __launch_bounds__(256)
final_kernel(const float* __restrict__ m2, const float* __restrict__ m3,
             float* __restrict__ comb, float* __restrict__ pred) {
  constexpr int HW = 784;
  __shared__ float sm[HW];
  const int b = blockIdx.x;
  for (int i = threadIdx.x; i < HW; i += 256) {
    const float v = 0.5f * (m2[b * HW + i] + m3[b * HW + i]);
    sm[i] = v;
    comb[b * HW + i] = v;
  }
  __syncthreads();
  if (threadIdx.x == 0) {
    float acc = 0.f;
    for (int k = 0; k < 10; ++k) {
      float bestv = -F_BIG;
      int   bi    = 0;
      for (int i = 0; i < HW; ++i)
        if (sm[i] > bestv) { bestv = sm[i]; bi = i; }
      acc += bestv;
      sm[bi] = -F_BIG;
    }
    pred[b] = acc * 0.1f;
  }
}

// -------------------------------------------------------------------------
extern "C" void kernel_launch(void* const* d_in, const int* in_sizes, int n_in,
                              void* d_out, int out_size, void* d_ws,
                              size_t ws_size, hipStream_t stream) {
  (void)in_sizes; (void)n_in; (void)out_size; (void)ws_size;

  constexpr int B = 16, HW = 784, N = B * HW;  // 12544
  constexpr int C2 = 128, C3 = 256, M = 8192;

  const float* q2    = (const float*)d_in[0];
  const float* q3    = (const float*)d_in[1];
  const float* mem2  = (const float*)d_in[2];
  const float* mem3  = (const float*)d_in[3];
  const float* mean2 = (const float*)d_in[4];
  const float* mean3 = (const float*)d_in[5];
  const float* icov2 = (const float*)d_in[6];
  const float* icov3 = (const float*)d_in[7];

  char* w = (char*)d_ws;
  _Float16* qh2 = (_Float16*)w;            w += (size_t)N * C2 * sizeof(_Float16);
  _Float16* qh3 = (_Float16*)w;            w += (size_t)N * C3 * sizeof(_Float16);
  _Float16* mh2 = (_Float16*)w;            w += (size_t)M * C2 * sizeof(_Float16);
  _Float16* mh3 = (_Float16*)w;            w += (size_t)M * C3 * sizeof(_Float16);
  float* qn2    = (float*)w;               w += (size_t)N * sizeof(float);
  float* qn3    = (float*)w;               w += (size_t)N * sizeof(float);
  float* mn2    = (float*)w;               w += (size_t)M * sizeof(float);
  float* mn3    = (float*)w;               w += (size_t)M * sizeof(float);
  float* maha2  = (float*)w;               w += (size_t)N * sizeof(float);
  float* maha3  = (float*)w;               w += (size_t)N * sizeof(float);

  float* out_pred = (float*)d_out;
  float* out_comb = out_pred + B;
  float* out_m2   = out_comb + N;
  float* out_m3   = out_m2 + N;

  prep_mem_kernel<C2><<<M / 8, 256, 0, stream>>>(mem2, mh2, mn2, M);
  prep_mem_kernel<C3><<<M / 8, 256, 0, stream>>>(mem3, mh3, mn3, M);

  prep_q_maha_kernel<C2><<<N / 32, 256, 0, stream>>>(q2, mean2, icov2, qh2, qn2, maha2, N, HW);
  prep_q_maha_kernel<C3><<<N / 32, 256, 0, stream>>>(q3, mean3, icov3, qh3, qn3, maha3, N, HW);

  knn_kernel<C2><<<N / 64, 128, 0, stream>>>(qh2, mh2, qn2, mn2, maha2, out_m2, M);
  knn_kernel<C3><<<N / 64, 128, 0, stream>>>(qh3, mh3, qn3, mn3, maha3, out_m3, M);

  final_kernel<<<B, 256, 0, stream>>>(out_m2, out_m3, out_comb, out_pred);
}
